// Memory_11682311045605
// MI455X (gfx1250) — matrix operand order repaired and checked by reading, not compiled, not run on hardware
//
#include <hip/hip_runtime.h>
#include <hip/hip_bf16.h>
#include <math.h>

// ---------------------------------------------------------------------------
// Problem constants (from setup_inputs):
//   B=16, L=4096, D_IN=512, D_H=1024, D_OUT=512, chunk=64, n_chunks=64
//   R = B*chunk = 1024 rows per chunk GEMM.
// Every GEMM in the pipeline has M,N in {512,1024}, K in {512,1024}:
// all multiples of 128/32 -> 128x128x32 WMMA tiles with no edge handling.
// ---------------------------------------------------------------------------
#define BATCH   16
#define SEQL    4096
#define DIN     512
#define DH      1024
#define DOUT    512
#define CHUNK   64
#define NCHUNK  64
#define RROWS   (BATCH * CHUNK)   // 1024

#define LR       0.1f
#define MOM      0.95f
#define NS_A     3.4445f
#define NS_B     (-4.7750f)
#define NS_C     2.0315f

typedef __attribute__((ext_vector_type(16))) _Float16 v16h;
typedef __attribute__((ext_vector_type(8)))  _Float16 v8hf;
typedef __attribute__((ext_vector_type(8)))  float    v8f;

// ---------------------------------------------------------------------------
// Tiled WMMA GEMM:  C(MxN, f32) = act(op(A) @ op(B)), A/B f16, optional f16
// mirror of C written to Ch.
//   op(A)[m][k] = TA ? A[k*lda + m] : A[m*lda + k]
//   op(B)[k][n] = TB ? B[n*ldb + k] : B[k*ldb + n]
// Block: 256 threads = 8 waves; BM=BN=128, BK=32. Wave grid 4x2; each wave
// computes 32x64 = 2x4 WMMA tiles -> 8 v_wmma per K-step per wave.
//
// LDS layouts are *fragment order* so each lane's v16h fragment is one
// contiguous 32-byte read (2 x ds_load_b128):
//   A: As[m][k]
//      lane fragment = As[m][lhi*16 .. lhi*16+15]
//   B: Bs[n][k]   (transposed during staging)
//      lane fragment = Bs[n][lhi*16 .. lhi*16+15]
//
// Software pipeline: double-buffered LDS. Next tile's global_load_b128 are
// issued into registers BEFORE the WMMA block (latency hidden behind matrix
// math), drained to the alternate LDS buffer after compute, one barrier per
// K-step. Requires M,N multiples of 128 and K multiple of 32.
// ---------------------------------------------------------------------------
#define BM 128
#define BN 128
#define BK 32
#define LROW 40   // BK + 8 pad halves; 80 bytes -> 16B-aligned vector slots

template <bool TA, bool TB>
__device__ __forceinline__ void stage_load(
    const _Float16* __restrict__ A, const _Float16* __restrict__ B,
    int blockRow, int blockCol, int k0, int lda, int ldb, int tid,
    v8hf av[2], v8hf bv[2])
{
    #pragma unroll
    for (int it = 0; it < 2; ++it) {
        int i = tid + it * 256;
        if (!TA) {
            int m = i >> 2, g = i & 3;
            av[it] = *(const v8hf*)&A[(size_t)(blockRow + m) * lda + k0 + g * 8];
        } else {
            int kk = i >> 4, mg = i & 15;
            av[it] = *(const v8hf*)&A[(size_t)(k0 + kk) * lda + blockRow + mg * 8];
        }
        if (TB) {
            int n = i >> 2, g = i & 3;
            bv[it] = *(const v8hf*)&B[(size_t)(blockCol + n) * ldb + k0 + g * 8];
        } else {
            int kk = i >> 4, ng = i & 15;
            bv[it] = *(const v8hf*)&B[(size_t)(k0 + kk) * ldb + blockCol + ng * 8];
        }
    }
}

template <bool TA, bool TB>
__device__ __forceinline__ void stage_store(
    _Float16 (*As)[LROW], _Float16 (*Bs)[LROW], int tid,
    const v8hf av[2], const v8hf bv[2])
{
    #pragma unroll
    for (int it = 0; it < 2; ++it) {
        int i = tid + it * 256;
        if (!TA) {
            int m = i >> 2, g = i & 3;
            int gp = g;   // contiguous k-groups
            *(v8hf*)&As[m][gp * 8] = av[it];
        } else {
            int kk = i >> 4, mg = i & 15;
            int ks = kk;
            #pragma unroll
            for (int j = 0; j < 8; ++j) As[mg * 8 + j][ks] = av[it][j];
        }
        if (TB) {
            int n = i >> 2, g = i & 3;
            *(v8hf*)&Bs[n][g * 8] = bv[it];
        } else {
            int kk = i >> 4, ng = i & 15;
            #pragma unroll
            for (int j = 0; j < 8; ++j) Bs[ng * 8 + j][kk] = bv[it][j];
        }
    }
}

template <bool TA, bool TB>
__global__ __launch_bounds__(256) void gemm_wmma_f16(
    const _Float16* __restrict__ A, const _Float16* __restrict__ B,
    float* __restrict__ C, _Float16* __restrict__ Ch, int act,
    int M, int N, int K, int lda, int ldb)
{
    __shared__ __align__(16) _Float16 As[2][BM][LROW];
    __shared__ __align__(16) _Float16 Bs[2][BN][LROW];

    const int tid  = threadIdx.x;
    const int wave = tid >> 5;            // 0..7
    const int lane = tid & 31;
    const int wm   = (wave >> 1) * 32;    // wave row offset (4 groups)
    const int wn   = (wave & 1) * 64;     // wave col offset (2 groups)
    const int blockRow = blockIdx.y * BM;
    const int blockCol = blockIdx.x * BN;

    const int lrow = lane & 15;           // 0..15
    const int lhi  = (lane >> 4) & 1;     // lane half select

    v8f acc[2][4];
    const v8f zacc = {0.f, 0.f, 0.f, 0.f, 0.f, 0.f, 0.f, 0.f};
    #pragma unroll
    for (int a = 0; a < 2; ++a)
        #pragma unroll
        for (int b = 0; b < 4; ++b) acc[a][b] = zacc;

    // ---- prologue: stage tile 0 into buffer 0 ----
    v8hf av[2], bv[2];
    stage_load<TA, TB>(A, B, blockRow, blockCol, 0, lda, ldb, tid, av, bv);
    stage_store<TA, TB>(As[0], Bs[0], tid, av, bv);
    __syncthreads();

    int cur = 0;
    for (int k0 = 0; k0 < K; k0 += BK) {
        const bool hasNext = (k0 + BK) < K;
        // issue next tile's global loads into registers (no wait yet)
        if (hasNext)
            stage_load<TA, TB>(A, B, blockRow, blockCol, k0 + BK,
                               lda, ldb, tid, av, bv);
        // prefetch tile after next (emits global_prefetch_b8)
        if (k0 + 2 * BK < K && tid == 0) {
            const _Float16* pf = TA ? &A[(size_t)(k0 + 2 * BK) * lda + blockRow]
                                    : &A[(size_t)blockRow * lda + (k0 + 2 * BK)];
            __builtin_prefetch((const void*)pf, 0, 1);
        }

        // ---- fragments: one contiguous 32B LDS read each ----
        v16h afr[2], bfr[4];
        #pragma unroll
        for (int tm = 0; tm < 2; ++tm)
            afr[tm] = *(const v16h*)&As[cur][wm + tm * 16 + lrow][lhi * 16];
        #pragma unroll
        for (int tn = 0; tn < 4; ++tn)
            bfr[tn] = *(const v16h*)&Bs[cur][wn + tn * 16 + lrow][lhi * 16];

        #pragma unroll
        for (int tm = 0; tm < 2; ++tm)
            #pragma unroll
            for (int tn = 0; tn < 4; ++tn)
                acc[tm][tn] = __builtin_amdgcn_wmma_f32_16x16x32_f16(
                    false, afr[tm], false, bfr[tn], (short)0, acc[tm][tn],
                    false, false);

        // drain next tile into the alternate LDS buffer, single barrier
        if (hasNext)
            stage_store<TA, TB>(As[cur ^ 1], Bs[cur ^ 1], tid, av, bv);
        __syncthreads();
        cur ^= 1;
    }

    // ---- epilogue: VGPR i holds M = i + lhi*8, N = lrow ----
    #pragma unroll
    for (int tm = 0; tm < 2; ++tm) {
        #pragma unroll
        for (int tn = 0; tn < 4; ++tn) {
            const int colg = blockCol + wn + tn * 16 + lrow;
            #pragma unroll
            for (int i = 0; i < 8; ++i) {
                const int rowg = blockRow + wm + tm * 16 + i + lhi * 8;
                float v = acc[tm][tn][i];
                if (act) v = tanhf(v);
                size_t idx = (size_t)rowg * N + colg;
                C[idx] = v;
                if (Ch) Ch[idx] = (_Float16)v;
            }
        }
    }
}

// ---------------------------------------------------------------------------
// Elementwise / small kernels
// ---------------------------------------------------------------------------
__global__ __launch_bounds__(256) void k_f32_to_f16(
    const float* __restrict__ s, _Float16* __restrict__ d, int n)
{
    int i = blockIdx.x * 256 + threadIdx.x;
    if (i < n) d[i] = (_Float16)s[i];
}

// gather chunk rows of K into dense f16 X (R x DIN), row r = b*CHUNK + t
__global__ __launch_bounds__(256) void k_gather_chunk(
    const float* __restrict__ kin, _Float16* __restrict__ xh, int chunk)
{
    int i = blockIdx.x * 256 + threadIdx.x;
    if (i >= RROWS * DIN) return;
    int r = i / DIN, d = i % DIN;
    int b = r / CHUNK, t = r % CHUNK;
    size_t src = ((size_t)b * SEQL + (size_t)chunk * CHUNK + t) * DIN + d;
    xh[i] = (_Float16)kin[src];
}

// dOUT = 2*(out - v)*mask / (B*CHUNK); mask from (b,t) tril band of width cw
__global__ __launch_bounds__(256) void k_dout(
    const float* __restrict__ out, const float* __restrict__ vin,
    _Float16* __restrict__ douth, const int* __restrict__ cwp, int chunk)
{
    int i = blockIdx.x * 256 + threadIdx.x;
    if (i >= RROWS * DOUT) return;
    int cw = *cwp;
    int r = i / DOUT, o = i % DOUT;
    int b = r / CHUNK, t = r % CHUNK;
    float mask = (t <= b && t > b - cw) ? 1.0f : 0.0f;
    size_t vsrc = ((size_t)b * SEQL + (size_t)chunk * CHUNK + t) * DOUT + o;
    float g = 2.0f * (out[i] - vin[vsrc]) * mask * (1.0f / (float)(BATCH * CHUNK));
    douth[i] = (_Float16)g;
}

// dpre = dH * (1 - H^2)  (f16 out); H is tanh'd activations (f32)
__global__ __launch_bounds__(256) void k_dpre(
    const float* __restrict__ dh, const float* __restrict__ h,
    _Float16* __restrict__ dpreh, int n)
{
    int i = blockIdx.x * 256 + threadIdx.x;
    if (i < n) {
        float hv = h[i];
        dpreh[i] = (_Float16)(dh[i] * (1.0f - hv * hv));
    }
}

// buf = MOM*buf + g ; t = g + MOM*buf (nesterov); thread 0 zeros norm scalar
__global__ __launch_bounds__(256) void k_muon_pre(
    const float* __restrict__ g, float* __restrict__ buf,
    float* __restrict__ t, float* __restrict__ scalar, int n)
{
    int i = blockIdx.x * 256 + threadIdx.x;
    if (i == 0) *scalar = 0.0f;
    if (i < n) {
        float bn = MOM * buf[i] + g[i];
        buf[i] = bn;
        t[i] = g[i] + MOM * bn;
    }
}

__global__ __launch_bounds__(256) void k_sqnorm(
    const float* __restrict__ t, float* __restrict__ scalar, int n)
{
    __shared__ float red[256];
    float s = 0.f;
    for (int i = blockIdx.x * 256 + threadIdx.x; i < n; i += gridDim.x * 256) {
        float v = t[i];
        s += v * v;
    }
    red[threadIdx.x] = s;
    __syncthreads();
    for (int o = 128; o > 0; o >>= 1) {
        if (threadIdx.x < o) red[threadIdx.x] += red[threadIdx.x + o];
        __syncthreads();
    }
    if (threadIdx.x == 0) atomicAdd(scalar, red[0]);
}

// X(512x1024) = (trans ? t^T : t) / (||t||_F + 1e-7); f32 + f16 mirrors
__global__ __launch_bounds__(256) void k_scale(
    const float* __restrict__ t, const float* __restrict__ scalar,
    float* __restrict__ x, _Float16* __restrict__ xh, int trans)
{
    int i = blockIdx.x * 256 + threadIdx.x;
    if (i >= DIN * DH) return;
    float inv = 1.0f / (sqrtf(*scalar) + 1e-7f);
    int m = i / DH, n = i % DH;
    float v = trans ? t[(size_t)n * DIN + m] : t[i];
    v *= inv;
    x[i] = v;
    xh[i] = (_Float16)v;
}

// Bm = NS_B*A + NS_C*A2  (f16 out)
__global__ __launch_bounds__(256) void k_bmat(
    const float* __restrict__ a, const float* __restrict__ a2,
    _Float16* __restrict__ bmh, int n)
{
    int i = blockIdx.x * 256 + threadIdx.x;
    if (i < n) bmh[i] = (_Float16)(NS_B * a[i] + NS_C * a2[i]);
}

// X = NS_A*X + Y ; refresh f16 mirror
__global__ __launch_bounds__(256) void k_axpy_x(
    float* __restrict__ x, const float* __restrict__ y,
    _Float16* __restrict__ xh, int n)
{
    int i = blockIdx.x * 256 + threadIdx.x;
    if (i < n) {
        float v = NS_A * x[i] + y[i];
        x[i] = v;
        xh[i] = (_Float16)v;
    }
}

// W -= LR * (trans ? X^T : X); refresh f16 mirror. W is rows x cols.
__global__ __launch_bounds__(256) void k_update(
    float* __restrict__ w, const float* __restrict__ x,
    _Float16* __restrict__ wh, int rows, int cols, int trans)
{
    int i = blockIdx.x * 256 + threadIdx.x;
    if (i >= rows * cols) return;
    float u;
    if (trans) {
        int r = i / cols, c = i % cols;        // W:1024x512, X:512x1024
        u = x[(size_t)c * DH + r];
    } else {
        u = x[i];                               // W:512x1024, same layout
    }
    float v = w[i] - LR * u;
    w[i] = v;
    wh[i] = (_Float16)v;
}

__global__ __launch_bounds__(256) void k_copy_f32(
    const float* __restrict__ s, float* __restrict__ d, int n)
{
    int i = blockIdx.x * 256 + threadIdx.x;
    if (i < n) d[i] = s[i];
}

// ---------------------------------------------------------------------------
// Host orchestration
// ---------------------------------------------------------------------------
static inline void launch_gemm(bool ta, bool tb,
                               const _Float16* A, const _Float16* B,
                               float* C, _Float16* Ch, int act,
                               int M, int N, int K, int lda, int ldb,
                               hipStream_t s)
{
    dim3 g(N / BN, M / BM), blk(256);
    if (!ta && !tb) gemm_wmma_f16<false, false><<<g, blk, 0, s>>>(A, B, C, Ch, act, M, N, K, lda, ldb);
    else if (!ta &&  tb) gemm_wmma_f16<false, true ><<<g, blk, 0, s>>>(A, B, C, Ch, act, M, N, K, lda, ldb);
    else if ( ta && !tb) gemm_wmma_f16<true , false><<<g, blk, 0, s>>>(A, B, C, Ch, act, M, N, K, lda, ldb);
    else                 gemm_wmma_f16<true , true ><<<g, blk, 0, s>>>(A, B, C, Ch, act, M, N, K, lda, ldb);
}

static inline int nb(int n) { return (n + 255) / 256; }

extern "C" void kernel_launch(void* const* d_in, const int* in_sizes, int n_in,
                              void* d_out, int out_size, void* d_ws, size_t ws_size,
                              hipStream_t stream)
{
    (void)in_sizes; (void)n_in; (void)out_size; (void)ws_size;
    const float* kin  = (const float*)d_in[0];
    const float* vin  = (const float*)d_in[1];
    const float* w1in = (const float*)d_in[2];
    const float* w2in = (const float*)d_in[3];
    const int*   cwp  = (const int*)d_in[4];
    float* outp = (float*)d_out;

    // ---- workspace bump allocator (256B aligned) ----
    size_t off = 0;
    auto alloc = [&](size_t bytes) -> void* {
        void* p = (char*)d_ws + off;
        off += (bytes + 255) & ~(size_t)255;
        return p;
    };
    float*    w1c    = (float*)alloc(sizeof(float) * DIN * DH);
    float*    w2c    = (float*)alloc(sizeof(float) * DH * DOUT);
    _Float16* w1h    = (_Float16*)alloc(sizeof(_Float16) * DIN * DH);
    _Float16* w2h    = (_Float16*)alloc(sizeof(_Float16) * DH * DOUT);
    _Float16* xch    = (_Float16*)alloc(sizeof(_Float16) * RROWS * DIN);
    float*    hf     = (float*)alloc(sizeof(float) * RROWS * DH);
    _Float16* hh     = (_Float16*)alloc(sizeof(_Float16) * RROWS * DH);
    float*    outf   = (float*)alloc(sizeof(float) * RROWS * DOUT);
    _Float16* douth  = (_Float16*)alloc(sizeof(_Float16) * RROWS * DOUT);
    float*    dhf    = (float*)alloc(sizeof(float) * RROWS * DH);
    _Float16* dpreh  = (_Float16*)alloc(sizeof(_Float16) * RROWS * DH);
    float*    g1     = (float*)alloc(sizeof(float) * DIN * DH);
    float*    g2     = (float*)alloc(sizeof(float) * DH * DOUT);
    float*    buf1   = (float*)alloc(sizeof(float) * DIN * DH);
    float*    buf2   = (float*)alloc(sizeof(float) * DH * DOUT);
    float*    tbuf   = (float*)alloc(sizeof(float) * DIN * DH);
    float*    xf     = (float*)alloc(sizeof(float) * DIN * DH);
    _Float16* xh2    = (_Float16*)alloc(sizeof(_Float16) * DIN * DH);
    float*    af     = (float*)alloc(sizeof(float) * DIN * DIN);
    _Float16* ah     = (_Float16*)alloc(sizeof(_Float16) * DIN * DIN);
    float*    a2f    = (float*)alloc(sizeof(float) * DIN * DIN);
    _Float16* bmh    = (_Float16*)alloc(sizeof(_Float16) * DIN * DIN);
    float*    yf     = (float*)alloc(sizeof(float) * DIN * DH);
    float*    nrm    = (float*)alloc(sizeof(float));

    // ---- init: working weight copies + f16 mirrors + zero momentum ----
    hipMemcpyAsync(w1c, w1in, sizeof(float) * DIN * DH, hipMemcpyDeviceToDevice, stream);
    hipMemcpyAsync(w2c, w2in, sizeof(float) * DH * DOUT, hipMemcpyDeviceToDevice, stream);
    hipMemsetAsync(buf1, 0, sizeof(float) * DIN * DH, stream);
    hipMemsetAsync(buf2, 0, sizeof(float) * DH * DOUT, stream);
    k_f32_to_f16<<<nb(DIN * DH), 256, 0, stream>>>(w1c, w1h, DIN * DH);
    k_f32_to_f16<<<nb(DH * DOUT), 256, 0, stream>>>(w2c, w2h, DH * DOUT);

    // Muon update sequence on one weight (NS5 on 512x1024 X)
    auto muon = [&](float* w, _Float16* wh, float* g, float* buf,
                    int rows, int cols, int trans) {
        const int n = rows * cols;        // == DIN*DH for both weights
        k_muon_pre<<<nb(n), 256, 0, stream>>>(g, buf, tbuf, nrm, n);
        k_sqnorm<<<128, 256, 0, stream>>>(tbuf, nrm, n);
        k_scale<<<nb(DIN * DH), 256, 0, stream>>>(tbuf, nrm, xf, xh2, trans);
        for (int it = 0; it < 5; ++it) {
            // A = X @ X^T : 512x512, K=1024; f16 mirror fused in epilogue
            launch_gemm(false, true, xh2, xh2, af, ah, 0, DIN, DIN, DH, DH, DH, stream);
            // A2 = A @ A : 512x512x512
            launch_gemm(false, false, ah, ah, a2f, nullptr, 0, DIN, DIN, DIN, DIN, DIN, stream);
            // Bm = b*A + c*A2
            k_bmat<<<nb(DIN * DIN), 256, 0, stream>>>(af, a2f, bmh, DIN * DIN);
            // Y = Bm @ X : 512x1024, K=512
            launch_gemm(false, false, bmh, xh2, yf, nullptr, 0, DIN, DH, DIN, DIN, DH, stream);
            // X = a*X + Y
            k_axpy_x<<<nb(DIN * DH), 256, 0, stream>>>(xf, yf, xh2, DIN * DH);
        }
        k_update<<<nb(n), 256, 0, stream>>>(w, xf, wh, rows, cols, trans);
    };

    // ---- sequential scan over chunks ----
    for (int c = 0; c < NCHUNK; ++c) {
        // X chunk -> f16
        k_gather_chunk<<<nb(RROWS * DIN), 256, 0, stream>>>(kin, xch, c);
        // H = tanh(X @ W1) (1024x1024, K=512), tanh + f16 mirror fused
        launch_gemm(false, false, xch, w1h, hf, hh, 1, RROWS, DH, DIN, DIN, DH, stream);
        // OUT = H @ W2 (1024x512, K=1024)
        launch_gemm(false, false, hh, w2h, outf, nullptr, 0, RROWS, DOUT, DH, DH, DOUT, stream);
        // dOUT
        k_dout<<<nb(RROWS * DOUT), 256, 0, stream>>>(outf, vin, douth, cwp, c);
        // G2 = H^T @ dOUT (1024x512, K=1024)
        launch_gemm(true, false, hh, douth, g2, nullptr, 0, DH, DOUT, RROWS, DH, DOUT, stream);
        // dH = dOUT @ W2^T (1024x1024, K=512)
        launch_gemm(false, true, douth, w2h, dhf, nullptr, 0, RROWS, DH, DOUT, DOUT, DOUT, stream);
        // dpre = dH * (1 - H^2)
        k_dpre<<<nb(RROWS * DH), 256, 0, stream>>>(dhf, hf, dpreh, RROWS * DH);
        // G1 = X^T @ dpre (512x1024, K=1024)
        launch_gemm(true, false, xch, dpreh, g1, nullptr, 0, DIN, DH, RROWS, DIN, DH, stream);
        // Muon updates
        muon(w1c, w1h, g1, buf1, DIN, DH, 0);   // 512x1024, no transpose
        muon(w2c, w2h, g2, buf2, DH, DOUT, 1);  // 1024x512, transposed in NS
    }

    // output = last chunk's OUT (pre-update weights of that chunk)
    k_copy_f32<<<nb(RROWS * DOUT), 256, 0, stream>>>(outf, outp, RROWS * DOUT);
}
